// HackableDecoderLayer_49838800503245
// MI455X (gfx1250) — matrix operand order repaired and checked by reading, not compiled, not run on hardware
//
#include <hip/hip_runtime.h>

// ---------------------------------------------------------------------------
// Decoder layer for MI455X (gfx1250), wave32 + v_wmma_f32_16x16x32_bf16,
// GEMM tiles staged via the Tensor Data Mover (TENSOR_LOAD_TO_LDS).
// B=2 S=2048 D=1024 H=16 KVH=8 HD=64 MLP=4096, fp32 in/out, bf16 matmul core.
// ---------------------------------------------------------------------------

#define B_    2
#define S_    2048
#define D_    1024
#define H_    16
#define KVH_  8
#define HD_   64
#define MLP_  4096
#define TOK_  (B_ * S_)          // 4096 rows
#define QKV_N (H_ * HD_ + 2 * KVH_ * HD_)  // 2048 fused projection width

typedef __attribute__((ext_vector_type(16))) __bf16 v16bf;
typedef __attribute__((ext_vector_type(8)))  __bf16 v8bf;
typedef __attribute__((ext_vector_type(8)))  float  v8f;

#define WMMA_BF16(a, b, c) \
  __builtin_amdgcn_wmma_f32_16x16x32_bf16(false, (a), false, (b), (short)0, (c), false, false)

__device__ __forceinline__ v16bf frag_combine(v8bf lo, v8bf hi) {
  return __builtin_shufflevector(lo, hi, 0, 1, 2, 3, 4, 5, 6, 7,
                                 8, 9, 10, 11, 12, 13, 14, 15);
}

// A-matrix 16x32 bf16 fragment: this lane's row = lane%16,
// elements K = {2*kb..2*kb+15} (contiguous), kb = (lane<16 ? 0 : 8).
__device__ __forceinline__ v16bf load_a_frag(const __bf16* rowptr, int kb) {
  v8bf lo = *(const v8bf*)(rowptr + 2 * kb);
  v8bf hi = *(const v8bf*)(rowptr + 2 * kb + 8);
  return frag_combine(lo, hi);
}

// ---------------------------------------------------------------------------
// Tensor Data Mover tile load (CDNA5). Builds a D# descriptor (ISA ch.8) for a
// dense 2D bf16 tile [tile_rows x tile_k] with row stride `stride_elems`,
// DMA'd into LDS packed row-major. Guarded: falls back to VGPR staging.
// This toolchain exposes the 6-arg builtin:
//   (uint32x4 g0, int32x8 g1, int32x4, int32x4, int32x8, i32 cpol)
// ---------------------------------------------------------------------------
#if __has_builtin(__builtin_amdgcn_tensor_load_to_lds) && \
    __has_builtin(__builtin_amdgcn_s_wait_tensorcnt)
#define USE_TDM 1
typedef unsigned int tdm_v4u __attribute__((ext_vector_type(4)));
typedef int          tdm_v8i __attribute__((ext_vector_type(8)));
typedef int          tdm_v4i __attribute__((ext_vector_type(4)));

__device__ __forceinline__ unsigned lds_byte_off(const void* p) {
  // Generic LDS pointer: addr[31:0] is the wave-relative LDS byte offset.
  return (unsigned)(uintptr_t)p;
}

__device__ __forceinline__ void tdm_load_tile_bf16(
    const __bf16* gsrc, unsigned lds_off,
    unsigned tile_k, unsigned tile_rows, unsigned stride_elems) {
  unsigned long long ga = (unsigned long long)(uintptr_t)gsrc;
  tdm_v4u g0;
  g0.x = 1u;                                         // count=1, dense load
  g0.y = lds_off;                                    // D#.lds_addr (bytes)
  g0.z = (unsigned)(ga & 0xFFFFFFFFull);             // global_addr[31:0]
  g0.w = (unsigned)((ga >> 32) & 0x01FFFFFFull)      // global_addr[56:32]
         | (2u << 30);                               // type=2 ("image")
  tdm_v8i g1;
  g1[0] = (int)(1u << 16);                           // data_size=1 (2 bytes)
  g1[1] = (int)((tile_k & 0xFFFFu) << 16);           // tensor_dim0[15:0]
  g1[2] = (int)((tile_k >> 16) |                     // tensor_dim0[31:16]
                ((tile_rows & 0xFFFFu) << 16));      // tensor_dim1[15:0]
  g1[3] = (int)((tile_rows >> 16) |                  // tensor_dim1[31:16]
                (tile_k << 16));                     // tile_dim0
  g1[4] = (int)(tile_rows & 0xFFFFu);                // tile_dim1 (tile_dim2=0)
  g1[5] = (int)stride_elems;                         // tensor_dim0_stride[31:0]
  g1[6] = 0;                                         // stride0 hi / stride1 lo
  g1[7] = 0;
  tdm_v4i z4 = {0, 0, 0, 0};                         // 2D: groups 2/3 unused
  tdm_v8i z8 = {0, 0, 0, 0, 0, 0, 0, 0};
  __builtin_amdgcn_tensor_load_to_lds(g0, g1, z4, z4, z8, 0);
}
#endif

// ---------------------------------------------------------------------------
// Elementwise: fp32 -> bf16 cast (weights)
// ---------------------------------------------------------------------------
__global__ __launch_bounds__(256) void cvt_f32_bf16_kernel(
    const float* __restrict__ in, __bf16* __restrict__ out, int n) {
  int i = blockIdx.x * 256 + threadIdx.x;
  if (i < n) out[i] = (__bf16)in[i];
}

// ---------------------------------------------------------------------------
// RMSNorm: one block per token row (D=1024, 256 thr x 4 elems) -> bf16
// ---------------------------------------------------------------------------
__global__ __launch_bounds__(256) void rmsnorm_kernel(
    const float* __restrict__ x, const float* __restrict__ w,
    __bf16* __restrict__ out) {
  const int row = blockIdx.x;
  const int tid = threadIdx.x;
  float4 v = ((const float4*)(x + (size_t)row * D_))[tid];
  float ss = v.x * v.x + v.y * v.y + v.z * v.z + v.w * v.w;
#pragma unroll
  for (int m = 16; m >= 1; m >>= 1) ss += __shfl_xor(ss, m, 32);
  __shared__ float red[8];
  if ((tid & 31) == 0) red[tid >> 5] = ss;
  __syncthreads();
  float tot = red[0] + red[1] + red[2] + red[3] +
              red[4] + red[5] + red[6] + red[7];
  float sc = rsqrtf(tot * (1.0f / D_) + 1e-5f);
  float4 wv = ((const float4*)w)[tid];
  __bf16* o = out + (size_t)row * D_ + tid * 4;
  o[0] = (__bf16)(v.x * sc * wv.x);
  o[1] = (__bf16)(v.y * sc * wv.y);
  o[2] = (__bf16)(v.z * sc * wv.z);
  o[3] = (__bf16)(v.w * sc * wv.w);
}

// ---------------------------------------------------------------------------
// Double-buffered bf16 WMMA GEMM:  C[M,N] = A[M,K] @ W[N,K]^T  (+ epilogue)
// Workgroup: 256 thr = 8 waves (4x2), tile 128x128; wave tile 32x64; K-step 32.
// Tiles are DMA'd global->LDS by the TDM (wave 0 issues, TENSORcnt-tracked),
// next tile in flight while WMMAs consume the current one.
// MODE 0: fp32 out    MODE 1: fp32 out + residual    MODE 3: silu(g)*u -> bf16
// ---------------------------------------------------------------------------
template <int MODE>
__global__ __launch_bounds__(256) void gemm_kernel(
    const __bf16* __restrict__ A, const __bf16* __restrict__ W,
    const __bf16* __restrict__ W2, const float* __restrict__ res,
    void* __restrict__ outp, int M, int N, int K) {
  __shared__ alignas(32) __bf16 lds_a[2][128 * 32];
  __shared__ alignas(32) __bf16 lds_w[2][128 * 32];
  __shared__ alignas(32) __bf16 lds_w2[2][128 * 32];

  const int tid = threadIdx.x;
  const int lane = tid & 31, wid = tid >> 5;
  const int half = lane >> 4, col = lane & 15;
  const int wm = wid >> 1, wn = wid & 1;
  const int m0 = blockIdx.y * 128, n0 = blockIdx.x * 128;

  v8f acc[2][4];
  v8f acc2[2][4];
#pragma unroll
  for (int i = 0; i < 2; ++i)
#pragma unroll
    for (int j = 0; j < 4; ++j) {
      acc[i][j] = (v8f){};
      acc2[i][j] = (v8f){};
    }

#if defined(USE_TDM)
  constexpr int NLOADS = (MODE == 3) ? 3 : 2;
  if (wid == 0) {  // one DMA issue per tile; TDM ignores EXEC, wave-level op
    tdm_load_tile_bf16(A + (size_t)m0 * K, lds_byte_off(lds_a[0]), 32, 128, K);
    tdm_load_tile_bf16(W + (size_t)n0 * K, lds_byte_off(lds_w[0]), 32, 128, K);
    if (MODE == 3)
      tdm_load_tile_bf16(W2 + (size_t)n0 * K, lds_byte_off(lds_w2[0]), 32, 128, K);
  }
#else
  const int lrow = tid >> 1;        // 0..127
  const int koff = (tid & 1) * 16;  // 0 or 16
  const __bf16* pa = A + (size_t)(m0 + lrow) * K + koff;
  const __bf16* pw = W + (size_t)(n0 + lrow) * K + koff;
  const __bf16* pw2 = (MODE == 3) ? (W2 + (size_t)(n0 + lrow) * K + koff) : pa;
  __bf16* const sa = lds_a[0] + lrow * 32 + koff;
  __bf16* const sw = lds_w[0] + lrow * 32 + koff;
  __bf16* const s2 = lds_w2[0] + lrow * 32 + koff;
  v8bf ra0 = ((const v8bf*)pa)[0], ra1 = ((const v8bf*)pa)[1];
  v8bf rw0 = ((const v8bf*)pw)[0], rw1 = ((const v8bf*)pw)[1];
  v8bf r20, r21;
  if (MODE == 3) { r20 = ((const v8bf*)pw2)[0]; r21 = ((const v8bf*)pw2)[1]; }
  ((v8bf*)sa)[0] = ra0; ((v8bf*)sa)[1] = ra1;
  ((v8bf*)sw)[0] = rw0; ((v8bf*)sw)[1] = rw1;
  if (MODE == 3) { ((v8bf*)s2)[0] = r20; ((v8bf*)s2)[1] = r21; }
#endif

  int buf = 0;
  for (int k0 = 0; k0 < K; k0 += 32) {
    const bool has_next = (k0 + 32) < K;

#if defined(USE_TDM)
    __syncthreads();  // all waves finished reading buffer buf^1 (prev iter)
    if (wid == 0) {
      if (has_next) {
        const int nb = buf ^ 1;
        tdm_load_tile_bf16(A + (size_t)m0 * K + k0 + 32,
                           lds_byte_off(lds_a[nb]), 32, 128, K);
        tdm_load_tile_bf16(W + (size_t)n0 * K + k0 + 32,
                           lds_byte_off(lds_w[nb]), 32, 128, K);
        if (MODE == 3)
          tdm_load_tile_bf16(W2 + (size_t)n0 * K + k0 + 32,
                             lds_byte_off(lds_w2[nb]), 32, 128, K);
        __builtin_amdgcn_s_wait_tensorcnt(NLOADS);  // current tile complete
      } else {
        __builtin_amdgcn_s_wait_tensorcnt(0);
      }
    }
    __syncthreads();  // publish buffer buf to all waves
#else
    if (has_next) {
      const v8bf* na = (const v8bf*)(pa + k0 + 32);
      const v8bf* nw = (const v8bf*)(pw + k0 + 32);
      ra0 = na[0]; ra1 = na[1];
      rw0 = nw[0]; rw1 = nw[1];
      if (MODE == 3) {
        const v8bf* n2 = (const v8bf*)(pw2 + k0 + 32);
        r20 = n2[0]; r21 = n2[1];
      }
    }
    __syncthreads();
#endif

    const __bf16* ba = lds_a[buf];
    const __bf16* bw = lds_w[buf];
    const __bf16* b2 = lds_w2[buf];
    v16bf af0 = load_a_frag(ba + (wm * 32 + 0 + col) * 32, half * 8);
    v16bf af1 = load_a_frag(ba + (wm * 32 + 16 + col) * 32, half * 8);
#pragma unroll
    for (int j = 0; j < 4; ++j) {
      v16bf wf = *(const v16bf*)(bw + (wn * 64 + j * 16 + col) * 32 + half * 16);
      acc[0][j] = WMMA_BF16(af0, wf, acc[0][j]);
      acc[1][j] = WMMA_BF16(af1, wf, acc[1][j]);
      if (MODE == 3) {
        v16bf wf2 = *(const v16bf*)(b2 + (wn * 64 + j * 16 + col) * 32 + half * 16);
        acc2[0][j] = WMMA_BF16(af0, wf2, acc2[0][j]);
        acc2[1][j] = WMMA_BF16(af1, wf2, acc2[1][j]);
      }
    }

#if !defined(USE_TDM)
    if (has_next) {
      const int o = (buf ^ 1) * 128 * 32;
      ((v8bf*)(sa + o))[0] = ra0; ((v8bf*)(sa + o))[1] = ra1;
      ((v8bf*)(sw + o))[0] = rw0; ((v8bf*)(sw + o))[1] = rw1;
      if (MODE == 3) { ((v8bf*)(s2 + o))[0] = r20; ((v8bf*)(s2 + o))[1] = r21; }
    }
#endif
    buf ^= 1;
  }

  // Epilogue. C layout: element r -> row = r + 8*half, col = lane%16.
#pragma unroll
  for (int i = 0; i < 2; ++i)
#pragma unroll
    for (int j = 0; j < 4; ++j)
#pragma unroll
      for (int r = 0; r < 8; ++r) {
        int row = m0 + wm * 32 + i * 16 + r + half * 8;
        int ccol = n0 + wn * 64 + j * 16 + col;
        size_t idx = (size_t)row * N + ccol;
        float g = acc[i][j][r];
        if (MODE == 0) {
          ((float*)outp)[idx] = g;
        } else if (MODE == 1) {
          ((float*)outp)[idx] = g + res[idx];
        } else {  // MODE 3: SwiGLU
          float u = acc2[i][j][r];
          float m = (g / (1.0f + __expf(-g))) * u;
          ((__bf16*)outp)[idx] = (__bf16)m;
        }
      }
}

// ---------------------------------------------------------------------------
// RoPE on Q slice of fused QKV fp32 [TOK, 2048] -> qb bf16 [B,S,H,HD]
// ---------------------------------------------------------------------------
__global__ __launch_bounds__(256) void rope_q_kernel(
    const float* __restrict__ qkv, __bf16* __restrict__ qb) {
  int idx = blockIdx.x * 256 + threadIdx.x;  // B*S*H*32 = 2^21 threads
  int i = idx & 31;
  int h = (idx >> 5) & (H_ - 1);
  int s = (idx >> 9) & (S_ - 1);
  int b = idx >> 20;
  size_t ib = (size_t)(b * S_ + s) * QKV_N + h * HD_;
  size_t ob = ((size_t)(b * S_ + s) * H_ + h) * HD_;
  float x1 = qkv[ib + i], x2 = qkv[ib + 32 + i];
  float freq = __powf(10000.0f, -(float)i / 32.0f);
  float sn, cs;
  __sincosf((float)s * freq, &sn, &cs);
  qb[ob + i] = (__bf16)(x1 * cs - x2 * sn);
  qb[ob + 32 + i] = (__bf16)(x2 * cs + x1 * sn);
}

// RoPE on K slice, permuting -> [B,KVH,S,HD] bf16
__global__ __launch_bounds__(256) void rope_k_kernel(
    const float* __restrict__ qkv, __bf16* __restrict__ kb) {
  int idx = blockIdx.x * 256 + threadIdx.x;  // B*S*KVH*32 = 2^20 threads
  int i = idx & 31;
  int kh = (idx >> 5) & (KVH_ - 1);
  int s = (idx >> 8) & (S_ - 1);
  int b = idx >> 19;
  size_t ib = (size_t)(b * S_ + s) * QKV_N + H_ * HD_ + kh * HD_;
  size_t ob = ((size_t)(b * KVH_ + kh) * S_ + s) * HD_;
  float x1 = qkv[ib + i], x2 = qkv[ib + 32 + i];
  float freq = __powf(10000.0f, -(float)i / 32.0f);
  float sn, cs;
  __sincosf((float)s * freq, &sn, &cs);
  kb[ob + i] = (__bf16)(x1 * cs - x2 * sn);
  kb[ob + 32 + i] = (__bf16)(x2 * cs + x1 * sn);
}

// V slice -> TRANSPOSED [B,KVH,HD,S] bf16 so the P@V B-fragment is a single
// contiguous 32B load (16 consecutive keys per lane).
__global__ __launch_bounds__(256) void cvt_v_kernel(
    const float* __restrict__ qkv, __bf16* __restrict__ vt) {
  int idx = blockIdx.x * 256 + threadIdx.x;  // 2^21 threads, s fastest
  int s = idx & (S_ - 1);
  int d = (idx >> 11) & (HD_ - 1);
  int kh = (idx >> 17) & (KVH_ - 1);
  int b = idx >> 20;
  size_t ib = (size_t)(b * S_ + s) * QKV_N + H_ * HD_ + KVH_ * HD_ + kh * HD_ + d;
  size_t ob = ((size_t)(b * KVH_ + kh) * HD_ + d) * S_ + s;
  vt[ob] = (__bf16)qkv[ib];
}

// ---------------------------------------------------------------------------
// Flash attention (causal, GQA rep=2). 128 thr = 4 waves; each wave owns a
// 16-row Q tile; KV blocks of 32; all fragments are contiguous 32B loads.
// Q: [B,S,H,HD] bf16.  K: [B,KVH,S,HD] bf16.  Vt: [B,KVH,HD,S] bf16.
// O: [B,S,H*HD] bf16.
// ---------------------------------------------------------------------------
__global__ __launch_bounds__(128) void attn_kernel(
    const __bf16* __restrict__ Q, const __bf16* __restrict__ Kt,
    const __bf16* __restrict__ Vt, __bf16* __restrict__ O) {
  __shared__ alignas(32) __bf16 lds_p[4][16 * 32];  // wave-private P tiles
  const int tid = threadIdx.x;
  const int lane = tid & 31, wid = tid >> 5;
  const int half = lane >> 4, col = lane & 15;
  const int h = blockIdx.y, b = blockIdx.z;
  const int q0 = blockIdx.x * 64 + wid * 16;
  const int kvh = h >> 1;

  const __bf16* qrow = Q + ((size_t)(b * S_ + q0 + col) * H_ + h) * HD_;
  v16bf qa0 = load_a_frag(qrow, half * 8);        // hd 0..31
  v16bf qa1 = load_a_frag(qrow + 32, half * 8);   // hd 32..63

  const __bf16* kbase = Kt + (size_t)(b * KVH_ + kvh) * S_ * HD_;
  const __bf16* vbase = Vt + (size_t)(b * KVH_ + kvh) * HD_ * S_;

  float mi[8], li[8];
  v8f o[4];
#pragma unroll
  for (int r = 0; r < 8; ++r) { mi[r] = -1e30f; li[r] = 0.0f; }
#pragma unroll
  for (int t = 0; t < 4; ++t) o[t] = (v8f){};

  const int kv_end = q0 + 15;
  for (int kv0 = 0; kv0 <= kv_end; kv0 += 32) {
    if (kv0 + 32 <= kv_end) {  // prefetch next KV block
      __builtin_prefetch(kbase + (size_t)(kv0 + 32 + col) * HD_, 0, 1);
      __builtin_prefetch(vbase + (size_t)col * S_ + kv0 + 32, 0, 1);
    }
    v8f st[2];
#pragma unroll
    for (int t = 0; t < 2; ++t) {
      int key = kv0 + t * 16 + col;
      const __bf16* krow = kbase + (size_t)key * HD_;
      v16bf kf0 = *(const v16bf*)(krow + half * 16);
      v16bf kf1 = *(const v16bf*)(krow + 32 + half * 16);
      v8f s = (v8f){};
      s = WMMA_BF16(qa0, kf0, s);
      s = WMMA_BF16(qa1, kf1, s);
#pragma unroll
      for (int r = 0; r < 8; ++r) {
        float sv = s[r] * 0.125f;  // 1/sqrt(64)
        int row = q0 + r + half * 8;
        if (key > row) sv = -1e30f;  // causal mask
        s[r] = sv;
      }
      st[t] = s;
    }
    // Online softmax; row stats replicated across the 16 lanes of a half.
    float al[8];
#pragma unroll
    for (int r = 0; r < 8; ++r) {
      float mx = fmaxf(st[0][r], st[1][r]);
#pragma unroll
      for (int msk = 8; msk >= 1; msk >>= 1)
        mx = fmaxf(mx, __shfl_xor(mx, msk, 32));
      float mn = fmaxf(mi[r], mx);
      float a = __expf(mi[r] - mn);
      float p0 = __expf(st[0][r] - mn);
      float p1 = __expf(st[1][r] - mn);
      st[0][r] = p0;
      st[1][r] = p1;
      float rs = p0 + p1;
#pragma unroll
      for (int msk = 8; msk >= 1; msk >>= 1) rs += __shfl_xor(rs, msk, 32);
      li[r] = li[r] * a + rs;
      mi[r] = mn;
      al[r] = a;
    }
#pragma unroll
    for (int t = 0; t < 4; ++t)
#pragma unroll
      for (int r = 0; r < 8; ++r) o[t][r] *= al[r];

    // Transpose P (accumulator layout) -> A-fragment layout via LDS bounce.
#pragma unroll
    for (int t = 0; t < 2; ++t)
#pragma unroll
      for (int r = 0; r < 8; ++r)
        lds_p[wid][(r + half * 8) * 32 + t * 16 + col] = (__bf16)st[t][r];
    v16bf pa = load_a_frag(&lds_p[wid][col * 32], half * 8);

    // P[16x32] @ V[32x64]: 4 WMMAs; Vt gives contiguous key runs per lane.
#pragma unroll
    for (int t = 0; t < 4; ++t) {
      v16bf vf = *(const v16bf*)(vbase + (size_t)(t * 16 + col) * S_ +
                                 kv0 + half * 16);
      o[t] = WMMA_BF16(pa, vf, o[t]);
    }
  }

  // Normalize and store O.
#pragma unroll
  for (int t = 0; t < 4; ++t)
#pragma unroll
    for (int r = 0; r < 8; ++r) {
      int srow = q0 + r + half * 8;
      float val = o[t][r] / li[r];
      O[(size_t)(b * S_ + srow) * (H_ * HD_) + h * HD_ + t * 16 + col] =
          (__bf16)val;
    }
}

// ---------------------------------------------------------------------------
// Host orchestration
// ---------------------------------------------------------------------------
extern "C" void kernel_launch(void* const* d_in, const int* in_sizes, int n_in,
                              void* d_out, int out_size, void* d_ws,
                              size_t ws_size, hipStream_t stream) {
  const float* x = (const float*)d_in[0];
  const float* w_ln1 = (const float*)d_in[1];
  const float* w_ln2 = (const float*)d_in[2];
  const float* Wq = (const float*)d_in[3];
  const float* Wk = (const float*)d_in[4];
  const float* Wv = (const float*)d_in[5];
  const float* Wo = (const float*)d_in[6];
  const float* Wg = (const float*)d_in[7];
  const float* Wu = (const float*)d_in[8];
  const float* Wd = (const float*)d_in[9];

  char* ws = (char*)d_ws;
  size_t off = 0;
  auto take = [&](size_t bytes) {
    char* p = ws + off;
    off = (off + bytes + 255) & ~(size_t)255;
    return p;
  };
  __bf16* hb     = (__bf16*)take((size_t)TOK_ * D_ * 2);        // rms1 out
  __bf16* wqkv_b = (__bf16*)take((size_t)QKV_N * D_ * 2);       // [Wq;Wk;Wv]
  __bf16* wo_b   = (__bf16*)take((size_t)D_ * H_ * HD_ * 2);
  __bf16* wg_b   = (__bf16*)take((size_t)MLP_ * D_ * 2);
  __bf16* wu_b   = (__bf16*)take((size_t)MLP_ * D_ * 2);
  __bf16* wd_b   = (__bf16*)take((size_t)D_ * MLP_ * 2);
  float*  qkvf   = (float*)take((size_t)TOK_ * QKV_N * 4);      // fused proj
  __bf16* qbb    = (__bf16*)take((size_t)TOK_ * H_ * HD_ * 2);
  __bf16* kbb    = (__bf16*)take((size_t)TOK_ * KVH_ * HD_ * 2);
  __bf16* vtb    = (__bf16*)take((size_t)TOK_ * KVH_ * HD_ * 2);
  __bf16* ob     = (__bf16*)take((size_t)TOK_ * H_ * HD_ * 2);
  float*  x1     = (float*)take((size_t)TOK_ * D_ * 4);         // attn residual
  __bf16* h2b    = (__bf16*)take((size_t)TOK_ * D_ * 2);
  __bf16* mb     = (__bf16*)take((size_t)TOK_ * MLP_ * 2);
  (void)ws_size; (void)n_in; (void)in_sizes; (void)out_size;

  auto cvt = [&](const float* src, __bf16* dst, int n) {
    cvt_f32_bf16_kernel<<<(n + 255) / 256, 256, 0, stream>>>(src, dst, n);
  };
  cvt(Wq, wqkv_b, H_ * HD_ * D_);
  cvt(Wk, wqkv_b + (size_t)H_ * HD_ * D_, KVH_ * HD_ * D_);
  cvt(Wv, wqkv_b + (size_t)(H_ + KVH_) * HD_ * D_, KVH_ * HD_ * D_);
  cvt(Wo, wo_b, D_ * H_ * HD_);
  cvt(Wg, wg_b, MLP_ * D_);
  cvt(Wu, wu_b, MLP_ * D_);
  cvt(Wd, wd_b, D_ * MLP_);

  // 1) h = rmsnorm(x, w_ln1)
  rmsnorm_kernel<<<TOK_, 256, 0, stream>>>(x, w_ln1, hb);

  // 2) fused QKV projection (fp32 out for RoPE)
  gemm_kernel<0><<<dim3(QKV_N / 128, TOK_ / 128), 256, 0, stream>>>(
      hb, wqkv_b, nullptr, nullptr, qkvf, TOK_, QKV_N, D_);

  // 3) RoPE + layout conversion to bf16 (V transposed for attention)
  rope_q_kernel<<<(B_ * S_ * H_ * 32) / 256, 256, 0, stream>>>(qkvf, qbb);
  rope_k_kernel<<<(B_ * S_ * KVH_ * 32) / 256, 256, 0, stream>>>(qkvf, kbb);
  cvt_v_kernel<<<(B_ * S_ * KVH_ * HD_) / 256, 256, 0, stream>>>(qkvf, vtb);

  // 4) causal GQA flash attention
  attn_kernel<<<dim3(S_ / 64, H_, B_), 128, 0, stream>>>(qbb, kbb, vtb, ob);

  // 5) x1 = x + o @ Wo^T
  gemm_kernel<1><<<dim3(D_ / 128, TOK_ / 128), 256, 0, stream>>>(
      ob, wo_b, nullptr, x, x1, TOK_, D_, H_ * HD_);

  // 6) h2 = rmsnorm(x1, w_ln2)
  rmsnorm_kernel<<<TOK_, 256, 0, stream>>>(x1, w_ln2, h2b);

  // 7) m = silu(h2 @ Wg^T) * (h2 @ Wu^T)  (fused gate+up GEMM)
  gemm_kernel<3><<<dim3(MLP_ / 128, TOK_ / 128), 256, 0, stream>>>(
      h2b, wg_b, wu_b, nullptr, mb, TOK_, MLP_, D_);

  // 8) out = x1 + m @ Wd^T
  gemm_kernel<1><<<dim3(D_ / 128, TOK_ / 128), 256, 0, stream>>>(
      mb, wd_b, nullptr, x1, (float*)d_out, TOK_, D_, MLP_);
}
